// MF2Net_32366873543134
// MI455X (gfx1250) — matrix operand order repaired and checked by reading, not compiled, not run on hardware
//
#include <hip/hip_runtime.h>
#include <math.h>

#define B_ROWS   262144
#define IN_DIM   512
#define HID      64
#define WAVES    8
#define PASSES   2          // per wave: 2 passes x 32 rows = 64 rows
#define BLOCK    256

// WMMA vector types (CDNA5 / gfx1250, wave32)
typedef __attribute__((ext_vector_type(16))) __bf16 v16bf;
typedef __attribute__((ext_vector_type(8)))  float  v8f;

union AF { v16bf v; __bf16 h[16]; unsigned short u[16]; };

__device__ __forceinline__ float sigmoidf_(float x) {
    return 1.0f / (1.0f + __expf(-x));
}

__device__ __forceinline__ float fuse_(float p0, float p1, float mu1, float mu2, float inc) {
    float mu12 = fminf(fmaxf(mu1, mu2) + inc, 1.0f);
    return (p0 <= p1) ? fmaf(p0, mu1, (p1 - p0) * mu12)
                      : fmaf(p1, mu2, (p0 - p1) * mu12);
}

// Build one 16x32 bf16 A fragment for this lane from two 32B row chunks.
__device__ __forceinline__ void packA_(AF& a, const float* __restrict__ xrow, int k0) {
    const float4* p = (const float4*)(xrow + k0);
    float4 f0 = p[0], f1 = p[1];
    const float4* q = (const float4*)(xrow + k0 + 16);
    float4 f2 = q[0], f3 = q[1];
    a.h[0]  = (__bf16)f0.x; a.h[1]  = (__bf16)f0.y;
    a.h[2]  = (__bf16)f0.z; a.h[3]  = (__bf16)f0.w;
    a.h[4]  = (__bf16)f1.x; a.h[5]  = (__bf16)f1.y;
    a.h[6]  = (__bf16)f1.z; a.h[7]  = (__bf16)f1.w;
    a.h[8]  = (__bf16)f2.x; a.h[9]  = (__bf16)f2.y;
    a.h[10] = (__bf16)f2.z; a.h[11] = (__bf16)f2.w;
    a.h[12] = (__bf16)f3.x; a.h[13] = (__bf16)f3.y;
    a.h[14] = (__bf16)f3.z; a.h[15] = (__bf16)f3.w;
}

// ---------------------------------------------------------------------------
// Prep: convert W1/W3 (512x64 f32, row-major [k][n]) into bf16 B-fragments
// laid out exactly as V_WMMA_F32_16X16X32_BF16 expects its B operand:
//   frag(mat,kt,nt): lane L, element e  <->  k = kt*32 + (e>=8)*16 + (L>=16)*8 + (e&7),
//                                            n = nt*16 + (L&15)
// Storage index == global thread index by construction.
// ---------------------------------------------------------------------------
__global__ void wprep_kernel(const float* __restrict__ W1,
                             const float* __restrict__ W3,
                             unsigned short* __restrict__ wsB) {
    int idx = blockIdx.x * blockDim.x + threadIdx.x;   // 0..65535
    int e  =  idx        & 15;
    int ln = (idx >> 4)  & 31;
    int nt = (idx >> 9)  & 3;
    int kt = (idx >> 11) & 15;
    int m  = (idx >> 15) & 1;
    int k = kt * 32 + (e & 8) * 2 + (ln >> 4) * 8 + (e & 7);
    int n = nt * 16 + (ln & 15);
    const float* W = m ? W3 : W1;
    AF tmp;
    tmp.h[0] = (__bf16)W[k * HID + n];
    wsB[idx] = tmp.u[0];
}

// ---------------------------------------------------------------------------
// Fused forward: h1=relu(x@W1+b1), h2=relu(x@W3+b3) via bf16 WMMA (f32 acc),
// 2 M-tiles (32 rows) per wave so every LDS B-fragment feeds two WMMAs,
// then 64x4 / 64x2 second layers via shuffle-reduced dots, sigmoid,
// probability fusion, 2-class softmax.
// ---------------------------------------------------------------------------
__global__ __launch_bounds__(BLOCK, 1)
void mf2net_fused_kernel(const float* __restrict__ probs,
                         const float* __restrict__ x,
                         const float* __restrict__ b1,
                         const float* __restrict__ W2,
                         const float* __restrict__ b2,
                         const float* __restrict__ b3,
                         const float* __restrict__ W4,
                         const float* __restrict__ b4,
                         const unsigned short* __restrict__ wsB,
                         float* __restrict__ out) {
    // All pre-swizzled B fragments for W1 and W3: 2*16*4 frags * 512 u16 = 128 KB
    __shared__ unsigned short sB[2 * 16 * 4 * 512];

    const int tid    = threadIdx.x;
    const int wave   = tid >> 5;
    const int lane   = tid & 31;
    const int laneM  = lane & 15;
    const int laneHi = lane >> 4;      // 0 or 1

    // Cooperative stage of all weight fragments into LDS (16B per thread/iter)
    {
        const uint4* src = (const uint4*)wsB;
        uint4*       dst = (uint4*)sB;
        #pragma unroll
        for (int i = 0; i < 32; ++i)
            dst[tid + i * BLOCK] = src[tid + i * BLOCK];
    }
    __syncthreads();

    // Hoist the tiny layer-2 weights / biases into registers (per-lane column)
    float b1v[4], b3v[4], w2v[4][4], w4v[4][2];
    #pragma unroll
    for (int nt = 0; nt < 4; ++nt) {
        int n = nt * 16 + laneM;
        b1v[nt] = b1[n];
        b3v[nt] = b3[n];
        float4 w2r = ((const float4*)W2)[n];   // W2[n][0..3]
        w2v[nt][0] = w2r.x; w2v[nt][1] = w2r.y;
        w2v[nt][2] = w2r.z; w2v[nt][3] = w2r.w;
        float2 w4r = ((const float2*)W4)[n];   // W4[n][0..1]
        w4v[nt][0] = w4r.x; w4v[nt][1] = w4r.y;
    }
    const float b2v0 = b2[0], b2v1 = b2[1], b2v2 = b2[2], b2v3 = b2[3];
    const float b4v0 = b4[0], b4v1 = b4[1];

    for (int t = 0; t < PASSES; ++t) {
        // 32 rows per pass: two 16-row M-tiles
        const int rowBase = (blockIdx.x * (WAVES * PASSES) + wave * PASSES + t) * 32;
        const float* xrow0 = x + (size_t)(rowBase + laneM) * IN_DIM + laneHi * 8;
        const float* xrow1 = xrow0 + (size_t)16 * IN_DIM;

        v8f acc1[2][4] = {};   // [m-tile][n-tile] h1 pre-activation
        v8f acc3[2][4] = {};   // [m-tile][n-tile] h2 pre-activation

        for (int kt = 0; kt < 16; ++kt) {
            AF a0, a1;
            packA_(a0, xrow0, kt * 32);
            packA_(a1, xrow1, kt * 32);

            #pragma unroll
            for (int nt = 0; nt < 4; ++nt) {
                const v16bf bA =
                    *(const v16bf*)(sB + ((0 * 16 + kt) * 4 + nt) * 512 + lane * 16);
                acc1[0][nt] = __builtin_amdgcn_wmma_f32_16x16x32_bf16(
                    false, a0.v, false, bA, (short)0, acc1[0][nt], false, false);
                acc1[1][nt] = __builtin_amdgcn_wmma_f32_16x16x32_bf16(
                    false, a1.v, false, bA, (short)0, acc1[1][nt], false, false);
                const v16bf bB =
                    *(const v16bf*)(sB + ((1 * 16 + kt) * 4 + nt) * 512 + lane * 16);
                acc3[0][nt] = __builtin_amdgcn_wmma_f32_16x16x32_bf16(
                    false, a0.v, false, bB, (short)0, acc3[0][nt], false, false);
                acc3[1][nt] = __builtin_amdgcn_wmma_f32_16x16x32_bf16(
                    false, a1.v, false, bB, (short)0, acc3[1][nt], false, false);
            }
        }

        // Epilogue per M-tile: lane holds cols {nt*16+laneM} of row (v + laneHi*8).
        // Butterfly-reduce the 4 partial cols across the 16-lane group to
        // complete the 64-wide dots for layer 2.
        #pragma unroll
        for (int mt = 0; mt < 2; ++mt) {
            const int tileRow = rowBase + mt * 16;
            #pragma unroll
            for (int v = 0; v < 8; ++v) {
                float t0 = 0.f, t1 = 0.f, t2 = 0.f, t3 = 0.f, u0 = 0.f, u1 = 0.f;
                #pragma unroll
                for (int nt = 0; nt < 4; ++nt) {
                    float h1n = fmaxf(acc1[mt][nt][v] + b1v[nt], 0.0f);
                    float h2n = fmaxf(acc3[mt][nt][v] + b3v[nt], 0.0f);
                    t0 = fmaf(h1n, w2v[nt][0], t0);
                    t1 = fmaf(h1n, w2v[nt][1], t1);
                    t2 = fmaf(h1n, w2v[nt][2], t2);
                    t3 = fmaf(h1n, w2v[nt][3], t3);
                    u0 = fmaf(h2n, w4v[nt][0], u0);
                    u1 = fmaf(h2n, w4v[nt][1], u1);
                }
                #pragma unroll
                for (int m = 1; m < 16; m <<= 1) {  // stays within 16-lane group
                    t0 += __shfl_xor(t0, m, 32);
                    t1 += __shfl_xor(t1, m, 32);
                    t2 += __shfl_xor(t2, m, 32);
                    t3 += __shfl_xor(t3, m, 32);
                    u0 += __shfl_xor(u0, m, 32);
                    u1 += __shfl_xor(u1, m, 32);
                }
                if (laneM == 0) {
                    const int R = tileRow + v + laneHi * 8;
                    float mu1c0 = sigmoidf_(t0 + b2v0);   // single[:,0,0]
                    float mu1c1 = sigmoidf_(t1 + b2v1);   // single[:,0,1]
                    float mu2c0 = sigmoidf_(t2 + b2v2);   // single[:,1,0]
                    float mu2c1 = sigmoidf_(t3 + b2v3);   // single[:,1,1]
                    float inc0  = sigmoidf_(u0 + b4v0);
                    float inc1  = sigmoidf_(u1 + b4v1);
                    float4 pr = ((const float4*)probs)[R]; // [p00,p01,p10,p11]
                    float r0 = fuse_(pr.x, pr.z, mu1c0, mu2c0, inc0);
                    float r1 = fuse_(pr.y, pr.w, mu1c1, mu2c1, inc1);
                    float mx = fmaxf(r0, r1);
                    float e0 = __expf(r0 - mx);
                    float e1 = __expf(r1 - mx);
                    float inv = 1.0f / (e0 + e1);
                    float2 o; o.x = e0 * inv; o.y = e1 * inv;
                    ((float2*)out)[R] = o;
                }
            }
        }
    }
}

extern "C" void kernel_launch(void* const* d_in, const int* in_sizes, int n_in,
                              void* d_out, int out_size, void* d_ws, size_t ws_size,
                              hipStream_t stream) {
    const float* probs = (const float*)d_in[0];
    const float* x     = (const float*)d_in[1];
    const float* W1    = (const float*)d_in[2];
    const float* b1    = (const float*)d_in[3];
    const float* W2    = (const float*)d_in[4];
    const float* b2    = (const float*)d_in[5];
    const float* W3    = (const float*)d_in[6];
    const float* b3    = (const float*)d_in[7];
    const float* W4    = (const float*)d_in[8];
    const float* b4    = (const float*)d_in[9];
    unsigned short* wsB = (unsigned short*)d_ws;   // needs 128 KB
    float* out = (float*)d_out;

    // 65536 fragment elements, one thread each
    wprep_kernel<<<256, 256, 0, stream>>>(W1, W3, wsB);

    const int grid = B_ROWS / (WAVES * PASSES * 32);   // 512 blocks
    mf2net_fused_kernel<<<grid, BLOCK, 0, stream>>>(
        probs, x, b1, W2, b2, b3, W4, b4, wsB, out);
}
